// NeuromorphicPrivacyNetwork_10144712753352
// MI455X (gfx1250) — matrix operand order repaired and checked by reading, not compiled, not run on hardware
//
#include <hip/hip_runtime.h>
#include <hip/hip_bf16.h>
#include <math.h>

// ---------------------------------------------------------------------------
// NeuromorphicPrivacyNetwork for MI455X (gfx1250), wave32 + WMMA bf16.
//   sizes: 1024 -> 2048 -> 1024 -> 512, T=16, B=256
//   - Layer-0 GEMM (inputs @ w0.T) is time-invariant: computed once.
//   - Spike GEMMs: v_wmma_f32_16x16x32_bf16, hi/lo bf16 weight split
//     (spikes are exact in bf16; hi+lo keeps ~16 mantissa bits of w).
//   - LDS block tiling (64x32 per workgroup) so A/B panels are read from L2
//     once per block instead of once per 16x16 tile.
//   - global->LDS staging uses the CDNA5 async DMA path
//     (GLOBAL_LOAD_ASYNC_TO_LDS_B128 + s_wait_asynccnt) with double-buffered
//     LDS, overlapping the next k-tile's DMA with the current tile's WMMAs
//     and eliminating the VGPR round-trip. Falls back to load+ds_store.
// ---------------------------------------------------------------------------

typedef __attribute__((ext_vector_type(16))) __bf16 v16bf;
typedef __attribute__((ext_vector_type(8)))  float  v8f;

#define LEAK        0.95f
#define REFR        2.0f
#define SYN_EPS     0.1f
#define NOISE_SCALE 0.05f

static constexpr int kT  = 16;
static constexpr int kB  = 256;
static constexpr int kN0 = 1024;   // input width
static constexpr int kN1 = 2048;   // layer0 out
static constexpr int kN2 = 1024;   // layer1 out
static constexpr int kN3 = 512;    // layer2 out

#if defined(__has_builtin)
#if __has_builtin(__builtin_amdgcn_global_load_async_to_lds_b128)
#define ASYNC_LDS 1
#endif
#endif
#ifndef ASYNC_LDS
#define ASYNC_LDS 0
#endif

#if ASYNC_LDS
#if __has_builtin(__builtin_amdgcn_s_wait_asynccnt)
#define WAIT_ASYNC(n) __builtin_amdgcn_s_wait_asynccnt(n)
#else
#define WAIT_ASYNC(n) asm volatile("s_wait_asynccnt %0" ::"i"(n) : "memory")
#endif
// builtin signature (from clang diagnostic): param0 = int4 addrspace(1)*,
// then LDS dest, imm offset, imm cpol.
typedef int v4i_ __attribute__((vector_size(16)));
typedef __attribute__((address_space(1))) v4i_ gv4i_t;   // global int4
typedef __attribute__((address_space(3))) v4i_ lv4i_t;   // LDS int4
// one 16-byte async DMA: global -> LDS, no VGPR round-trip, ASYNCcnt-tracked
__device__ __forceinline__ void async_cp16(const __bf16* g, __bf16* l) {
  __builtin_amdgcn_global_load_async_to_lds_b128((gv4i_t*)g, (lv4i_t*)l, 0, 0);
}
#endif

union F8 { uint4 u; __bf16 h[8]; };

// A fragment (16x32 bf16, M x K) from row-major [ld] storage:
//   lane<16 -> row base+lane, K = {0..7, 16..23}; lane>=16 -> K = {8..15, 24..31}
__device__ __forceinline__ v16bf load_frag_a(const __bf16* base, int ld, int row,
                                             int half) {
  const __bf16* p = base + (size_t)row * ld + half * 8;
  F8 x0, x1;
  x0.u = *(const uint4*)(p);
  x1.u = *(const uint4*)(p + 16);
  v16bf a;
#pragma unroll
  for (int i = 0; i < 8; ++i) { a[i] = x0.h[i]; a[i + 8] = x1.h[i]; }
  return a;
}

// B fragment (32x16 bf16, K x N) from row-major W[N][K] (B = W^T):
//   lane<16 -> col base+lane, K = {0..7, 16..23}; lane>=16 -> K = {8..15, 24..31}
__device__ __forceinline__ v16bf load_frag_b(const __bf16* base, int ld, int col,
                                             int half) {
  const __bf16* p = base + (size_t)col * ld + half * 8;
  F8 x0, x1;
  x0.u = *(const uint4*)(p);
  x1.u = *(const uint4*)(p + 16);
  v16bf b;
#pragma unroll
  for (int i = 0; i < 8; ++i) { b[i] = x0.h[i]; b[i + 8] = x1.h[i]; }
  return b;
}

// ------------------------------- prep kernels ------------------------------

__global__ __launch_bounds__(256) void k_split_bf16(const float* __restrict__ src,
                                                    __bf16* __restrict__ hi,
                                                    __bf16* __restrict__ lo, int n) {
  int i = blockIdx.x * blockDim.x + threadIdx.x;
  if (i < n) {
    float v  = src[i];
    __bf16 h = (__bf16)v;
    hi[i] = h;
    lo[i] = (__bf16)(v - (float)h);
  }
}

__global__ __launch_bounds__(256) void k_wnorm(const float* __restrict__ w,
                                               float* __restrict__ wn, int K) {
  int row = blockIdx.x;
  const float* p = w + (size_t)row * K;
  float s = 0.f;
  for (int k = threadIdx.x; k < K; k += 256) { float v = p[k]; s += v * v; }
  __shared__ float red[256];
  red[threadIdx.x] = s;
  __syncthreads();
  for (int off = 128; off > 0; off >>= 1) {
    if (threadIdx.x < off) red[threadIdx.x] += red[threadIdx.x + off];
    __syncthreads();
  }
  if (threadIdx.x == 0) wn[row] = sqrtf(red[0]);
}

__global__ __launch_bounds__(256) void k_init_state(float* __restrict__ V,
                                                    float* __restrict__ ls, int n) {
  int i = blockIdx.x * blockDim.x + threadIdx.x;
  if (i < n) { V[i] = 0.f; ls[i] = -1e9f; }
}

__global__ void k_zero_acc(float* acc) {
  if (threadIdx.x == 0 && blockIdx.x == 0) { acc[0] = 0.f; acc[1] = 0.f; }
}

// ---------------- LDS-tiled WMMA GEMM (+ optional LIF epilogue) ------------
// Block tile: BM=64 x BN=32, 8 waves, each wave owns one 16x16 subtile.
// A_SPLIT: A has hi+lo planes (f32 x f32 GEMM via 3 bf16 products).
// DO_LIF : fused leaky-integrate-and-fire epilogue + spike/cost stats.
template <bool A_SPLIT, bool DO_LIF>
__global__ __launch_bounds__(256) void k_gemm_tiled(
    const __bf16* __restrict__ Ahi, const __bf16* __restrict__ Alo,
    const __bf16* __restrict__ Bhi, const __bf16* __restrict__ Blo,
    float* __restrict__ Zout,
    const float* __restrict__ cn_t, const float* __restrict__ tn_t,
    const float* __restrict__ th, const float* __restrict__ wn,
    float* __restrict__ V, float* __restrict__ ls,
    __bf16* __restrict__ sp_out, float* __restrict__ out_f32,
    float* __restrict__ acc, float tval, float costw,
    int M, int N, int K) {
  constexpr int BM = 64, BN = 32, BK = 32;
  constexpr int APL = A_SPLIT ? 2 : 1;

  int tid  = threadIdx.x;
  int wave = tid >> 5;
  int lane = tid & 31;
  int lr   = lane & 15;
  int half = lane >> 4;
  int msub = wave & 3;        // 4 m-subtiles
  int nsub = wave >> 2;       // 2 n-subtiles

  int bm = blockIdx.x % (M / BM);
  int bn = blockIdx.x / (M / BM);
  int m0 = bm * BM;
  int n0 = bn * BN;

  // cooperative-load assignments (uniform per wave-pair)
  int arow = tid >> 2, aq = (tid & 3) * 8;                    // A: 64x32, 16B/thr/plane
  int bplane = tid >> 7;                                      // waves 0-3: hi, 4-7: lo
  int bt = tid & 127;
  int brow = bt >> 2, bq = (bt & 3) * 8;                      // B: 32x32 per plane

  v8f c = {};

#if ASYNC_LDS
  // -------- async DMA staging, double-buffered LDS --------
  __shared__ __bf16 lA[2][APL * BM * BK];
  __shared__ __bf16 lB[2][2 * BN * BK];

  auto stage = [&](int buf, int k0) {
    async_cp16(Ahi + (size_t)(m0 + arow) * K + k0 + aq, &lA[buf][arow * BK + aq]);
    if (A_SPLIT)
      async_cp16(Alo + (size_t)(m0 + arow) * K + k0 + aq,
                 &lA[buf][BM * BK + arow * BK + aq]);
    const __bf16* bsrc = bplane ? Blo : Bhi;
    async_cp16(bsrc + (size_t)(n0 + brow) * K + k0 + bq,
               &lB[buf][bplane * BN * BK + brow * BK + bq]);
  };

  stage(0, 0);
  int nk = K / BK;
  for (int i = 0; i < nk; ++i) {
    int  cur = i & 1;
    bool pre = (i + 1) < nk;
    if (pre) stage(cur ^ 1, (i + 1) * BK);
    // async loads retire in order: waiting <= (APL+1) retires this buffer's
    // transfers while next buffer's remain in flight.
    if (pre) WAIT_ASYNC(APL + 1); else WAIT_ASYNC(0);
    __syncthreads();

    v16bf ah = load_frag_a(&lA[cur][0], BK, msub * 16 + lr, half);
    v16bf bh = load_frag_b(&lB[cur][0], BK, nsub * 16 + lr, half);
    v16bf bl = load_frag_b(&lB[cur][BN * BK], BK, nsub * 16 + lr, half);
    c = __builtin_amdgcn_wmma_f32_16x16x32_bf16(false, ah, false, bh, (short)0, c, false, false);
    c = __builtin_amdgcn_wmma_f32_16x16x32_bf16(false, ah, false, bl, (short)0, c, false, false);
    if (A_SPLIT) {
      v16bf al = load_frag_a(&lA[cur][BM * BK], BK, msub * 16 + lr, half);
      c = __builtin_amdgcn_wmma_f32_16x16x32_bf16(false, al, false, bh, (short)0, c, false, false);
    }
    __syncthreads();  // all ds_loads consumed; safe to DMA over this buffer
  }
#else
  // -------- fallback: load-to-VGPR + ds_store staging --------
  __shared__ __bf16 lA1[APL * BM * BK];
  __shared__ __bf16 lB1[2 * BN * BK];

  for (int k0 = 0; k0 < K; k0 += BK) {
    *(uint4*)(lA1 + arow * BK + aq) =
        *(const uint4*)(Ahi + (size_t)(m0 + arow) * K + k0 + aq);
    if (A_SPLIT) {
      *(uint4*)(lA1 + BM * BK + arow * BK + aq) =
          *(const uint4*)(Alo + (size_t)(m0 + arow) * K + k0 + aq);
    }
    const __bf16* bsrc = bplane ? Blo : Bhi;
    *(uint4*)(lB1 + bplane * BN * BK + brow * BK + bq) =
        *(const uint4*)(bsrc + (size_t)(n0 + brow) * K + k0 + bq);
    __syncthreads();

    v16bf ah = load_frag_a(lA1, BK, msub * 16 + lr, half);
    v16bf bh = load_frag_b(lB1, BK, nsub * 16 + lr, half);
    v16bf bl = load_frag_b(lB1 + BN * BK, BK, nsub * 16 + lr, half);
    c = __builtin_amdgcn_wmma_f32_16x16x32_bf16(false, ah, false, bh, (short)0, c, false, false);
    c = __builtin_amdgcn_wmma_f32_16x16x32_bf16(false, ah, false, bl, (short)0, c, false, false);
    if (A_SPLIT) {
      v16bf al = load_frag_a(lA1 + BM * BK, BK, msub * 16 + lr, half);
      c = __builtin_amdgcn_wmma_f32_16x16x32_bf16(false, al, false, bh, (short)0, c, false, false);
    }
    __syncthreads();
  }
#endif

  // D layout: lane half selects M base 0/8; c[r] = (mb + r, n)
  int mb = m0 + msub * 16 + half * 8;
  int n  = n0 + nsub * 16 + lr;

  if (!DO_LIF) {
#pragma unroll
    for (int r = 0; r < 8; ++r) Zout[(size_t)(mb + r) * N + n] = c[r];
    return;
  }

  float wnn = wn[n];
  float thn = th[n];
  float myspk = 0.f, mycost = 0.f;
#pragma unroll
  for (int r = 0; r < 8; ++r) {
    size_t idx = (size_t)(mb + r) * N + n;
    float cur  = c[r] + cn_t[idx] * (NOISE_SCALE * wnn);
    float Vv   = V[idx];
    float L    = ls[idx];
    bool refr  = (tval - L) < REFR;
    float Vn   = refr ? Vv : (LEAK * Vv + cur);
    float thr  = thn + tn_t[idx] * SYN_EPS;
    bool sp    = (!refr) && (Vn > thr);
    V[idx]     = sp ? 0.f : Vn;
    ls[idx]    = sp ? tval : L;
    sp_out[idx] = sp ? (__bf16)1.0f : (__bf16)0.0f;
    if (out_f32) out_f32[idx] = sp ? 1.f : 0.f;
    myspk  += sp ? 1.f : 0.f;
    mycost += sp ? wnn : 0.f;
  }
  __shared__ float rs[256], rc[256];
  rs[tid] = myspk;
  rc[tid] = mycost;
  __syncthreads();
  for (int off = 128; off > 0; off >>= 1) {
    if (tid < off) { rs[tid] += rs[tid + off]; rc[tid] += rc[tid + off]; }
    __syncthreads();
  }
  if (tid == 0) {
    atomicAdd(&acc[0], costw * rc[0]);
    atomicAdd(&acc[1], rs[0]);
  }
}

// ----------------- layer-0 LIF (elementwise, Z0 precomputed) ---------------
__global__ __launch_bounds__(256) void k_lif0(const float* __restrict__ Z,
                                              const float* __restrict__ cn_t,
                                              const float* __restrict__ tn_t,
                                              const float* __restrict__ th,
                                              const float* __restrict__ wn,
                                              float* __restrict__ V,
                                              float* __restrict__ ls,
                                              __bf16* __restrict__ sp_out,
                                              float* __restrict__ acc,
                                              float tval, float costw, int M, int N) {
  int i = blockIdx.x * blockDim.x + threadIdx.x;
  float myspk = 0.f, mycost = 0.f;
  if (i < M * N) {
    int   n   = i & (N - 1);                       // N is a power of two
    float wnn = wn[n];
    float cur = Z[i] + cn_t[i] * (NOISE_SCALE * wnn);
    float Vv  = V[i];
    float L   = ls[i];
    bool refr = (tval - L) < REFR;
    float Vn  = refr ? Vv : (LEAK * Vv + cur);
    float thr = th[n] + tn_t[i] * SYN_EPS;
    bool sp   = (!refr) && (Vn > thr);
    V[i]      = sp ? 0.f : Vn;
    ls[i]     = sp ? tval : L;
    sp_out[i] = sp ? (__bf16)1.0f : (__bf16)0.0f;
    myspk     = sp ? 1.f : 0.f;
    mycost    = sp ? wnn : 0.f;
  }
  __shared__ float rs[256], rc[256];
  rs[threadIdx.x] = myspk;
  rc[threadIdx.x] = mycost;
  __syncthreads();
  for (int off = 128; off > 0; off >>= 1) {
    if (threadIdx.x < off) {
      rs[threadIdx.x] += rs[threadIdx.x + off];
      rc[threadIdx.x] += rc[threadIdx.x + off];
    }
    __syncthreads();
  }
  if (threadIdx.x == 0) {
    atomicAdd(&acc[0], costw * rc[0]);
    atomicAdd(&acc[1], rs[0]);
  }
}

// ------------------------------- finalize ----------------------------------
__global__ void k_finalize(const float* __restrict__ acc, float* __restrict__ tail) {
  if (threadIdx.x == 0 && blockIdx.x == 0) {
    const float cnt = (float)kT * (float)kB * (float)(kN1 + kN2 + kN3);
    float p1 = acc[1] / cnt;
    float p0 = 1.f - p1;
    const float e = 1e-12f;
    float ent = -(p1 * log2f(p1 + e) + p0 * log2f(p0 + e));
    tail[0] = acc[0];   // privacy cost
    tail[1] = ent;      // entropy
  }
}

// ------------------------------- launcher ----------------------------------
static inline char* carve(char*& p, size_t bytes) {
  char* r = p;
  p += (bytes + 255) & ~(size_t)255;
  return r;
}

extern "C" void kernel_launch(void* const* d_in, const int* in_sizes, int n_in,
                              void* d_out, int out_size, void* d_ws, size_t ws_size,
                              hipStream_t stream) {
  (void)in_sizes; (void)n_in; (void)out_size; (void)ws_size;
  const float* x   = (const float*)d_in[0];
  const float* w0  = (const float*)d_in[1];
  const float* th0 = (const float*)d_in[2];
  const float* cn0 = (const float*)d_in[3];
  const float* tn0 = (const float*)d_in[4];
  const float* w1  = (const float*)d_in[5];
  const float* th1 = (const float*)d_in[6];
  const float* cn1 = (const float*)d_in[7];
  const float* tn1 = (const float*)d_in[8];
  const float* w2  = (const float*)d_in[9];
  const float* th2 = (const float*)d_in[10];
  const float* cn2 = (const float*)d_in[11];
  const float* tn2 = (const float*)d_in[12];
  // d_in[13] = time_steps == 16 (compile-time constant here)

  // workspace carve
  char* p = (char*)d_ws;
  __bf16* xhi  = (__bf16*)carve(p, (size_t)kB  * kN0 * 2);
  __bf16* xlo  = (__bf16*)carve(p, (size_t)kB  * kN0 * 2);
  __bf16* w0hi = (__bf16*)carve(p, (size_t)kN1 * kN0 * 2);
  __bf16* w0lo = (__bf16*)carve(p, (size_t)kN1 * kN0 * 2);
  __bf16* w1hi = (__bf16*)carve(p, (size_t)kN2 * kN1 * 2);
  __bf16* w1lo = (__bf16*)carve(p, (size_t)kN2 * kN1 * 2);
  __bf16* w2hi = (__bf16*)carve(p, (size_t)kN3 * kN2 * 2);
  __bf16* w2lo = (__bf16*)carve(p, (size_t)kN3 * kN2 * 2);
  float*  wn0  = (float*)carve(p, (size_t)kN1 * 4);
  float*  wn1  = (float*)carve(p, (size_t)kN2 * 4);
  float*  wn2  = (float*)carve(p, (size_t)kN3 * 4);
  float*  Z0   = (float*)carve(p, (size_t)kB * kN1 * 4);
  float*  V0   = (float*)carve(p, (size_t)kB * kN1 * 4);
  float*  ls0  = (float*)carve(p, (size_t)kB * kN1 * 4);
  float*  V1   = (float*)carve(p, (size_t)kB * kN2 * 4);
  float*  ls1  = (float*)carve(p, (size_t)kB * kN2 * 4);
  float*  V2   = (float*)carve(p, (size_t)kB * kN3 * 4);
  float*  ls2  = (float*)carve(p, (size_t)kB * kN3 * 4);
  __bf16* sp0  = (__bf16*)carve(p, (size_t)kB * kN1 * 2);
  __bf16* sp1  = (__bf16*)carve(p, (size_t)kB * kN2 * 2);
  __bf16* sp2  = (__bf16*)carve(p, (size_t)kB * kN3 * 2);
  float*  acc  = (float*)carve(p, 2 * 4);

  dim3 blk(256);

  // prep: bf16 hi/lo splits, per-neuron weight norms, state init
  k_split_bf16<<<(kB  * kN0) / 256, blk, 0, stream>>>(x,  xhi,  xlo,  kB  * kN0);
  k_split_bf16<<<(kN1 * kN0) / 256, blk, 0, stream>>>(w0, w0hi, w0lo, kN1 * kN0);
  k_split_bf16<<<(kN2 * kN1) / 256, blk, 0, stream>>>(w1, w1hi, w1lo, kN2 * kN1);
  k_split_bf16<<<(kN3 * kN2) / 256, blk, 0, stream>>>(w2, w2hi, w2lo, kN3 * kN2);
  k_wnorm<<<kN1, blk, 0, stream>>>(w0, wn0, kN0);
  k_wnorm<<<kN2, blk, 0, stream>>>(w1, wn1, kN1);
  k_wnorm<<<kN3, blk, 0, stream>>>(w2, wn2, kN2);
  k_init_state<<<(kB * kN1) / 256, blk, 0, stream>>>(V0, ls0, kB * kN1);
  k_init_state<<<(kB * kN2) / 256, blk, 0, stream>>>(V1, ls1, kB * kN2);
  k_init_state<<<(kB * kN3) / 256, blk, 0, stream>>>(V2, ls2, kB * kN3);
  k_zero_acc<<<1, 32, 0, stream>>>(acc);

  // time-invariant layer-0 GEMM: Z0 = inputs @ w0.T   [256, 2048]
  k_gemm_tiled<true, false><<<(kB / 64) * (kN1 / 32), blk, 0, stream>>>(
      xhi, xlo, w0hi, w0lo, Z0,
      nullptr, nullptr, nullptr, nullptr, nullptr, nullptr, nullptr, nullptr,
      nullptr, 0.f, 0.f, kB, kN1, kN0);

  for (int t = 1; t <= kT; ++t) {
    float tv = (float)t;
    int   wwin  = (17 - t) < 10 ? (17 - t) : 10;   // trailing-window multiplicity
    float costw = SYN_EPS * (float)wwin / 10.0f;   // = 0.01 * min(10, 17-t)
    const float* cn0t = cn0 + (size_t)(t - 1) * kB * kN1;
    const float* tn0t = tn0 + (size_t)(t - 1) * kB * kN1;
    const float* cn1t = cn1 + (size_t)(t - 1) * kB * kN2;
    const float* tn1t = tn1 + (size_t)(t - 1) * kB * kN2;
    const float* cn2t = cn2 + (size_t)(t - 1) * kB * kN3;
    const float* tn2t = tn2 + (size_t)(t - 1) * kB * kN3;

    k_lif0<<<(kB * kN1) / 256, blk, 0, stream>>>(
        Z0, cn0t, tn0t, th0, wn0, V0, ls0, sp0, acc, tv, costw, kB, kN1);

    k_gemm_tiled<false, true><<<(kB / 64) * (kN2 / 32), blk, 0, stream>>>(
        sp0, nullptr, w1hi, w1lo, nullptr,
        cn1t, tn1t, th1, wn1, V1, ls1, sp1, (float*)nullptr,
        acc, tv, costw, kB, kN2, kN1);

    float* outp = (t == kT) ? (float*)d_out : (float*)nullptr;
    k_gemm_tiled<false, true><<<(kB / 64) * (kN3 / 32), blk, 0, stream>>>(
        sp1, nullptr, w2hi, w2lo, nullptr,
        cn2t, tn2t, th2, wn2, V2, ls2, sp2, outp,
        acc, tv, costw, kB, kN3, kN2);
  }

  k_finalize<<<1, 32, 0, stream>>>(acc, (float*)d_out + (size_t)kB * kN3);
}
